// SocialAttention_88562225644177
// MI455X (gfx1250) — matrix operand order repaired and checked by reading, not compiled, not run on hardware
//
#include <hip/hip_runtime.h>

// ---------------------------------------------------------------------------
// SocialAttention on MI455X (gfx1250)
//   q = relu(enc @ Wq + bq) / 16          [B,256]   (1/sqrt(256) = 0.0625)
//   k = relu(soc @ Wk + bk)               [N,256]
//   v = relu(soc @ Wv + bv)               [N,256]
//   out[i] = softmax(q_i . k[start_i:end_i]) @ v[start_i:end_i]
//
// Projections (8.8 GFLOP dense) -> bf16 WMMA GEMM, A tiles staged into LDS
// via the Tensor Data Mover (tensor_load_to_lds + s_wait_tensorcnt).
// Attention: ragged <=64-wide contiguous windows -> VALU + wave32 shuffles
// (avoids the reference's full BxN score matrix: -17 GFLOP, -128 MB traffic).
// ---------------------------------------------------------------------------

#define B_SZ 1024
#define N_SZ 32768
#define H_SZ 256
#define MAXW 64

typedef __attribute__((ext_vector_type(4)))  float          v4f;
typedef __attribute__((ext_vector_type(8)))  float          v8f;
typedef __attribute__((ext_vector_type(16))) __bf16         v16bf;
typedef __attribute__((ext_vector_type(4)))  unsigned short u16x4;
typedef __attribute__((ext_vector_type(8)))  unsigned short u16x8;
typedef __attribute__((ext_vector_type(16))) unsigned short u16x16;
typedef __attribute__((ext_vector_type(4)))  unsigned int   u32x4;
typedef __attribute__((ext_vector_type(4)))  int            i32x4;
typedef __attribute__((ext_vector_type(8)))  int            i32x8;

#if defined(__has_builtin)
#  if __has_builtin(__builtin_amdgcn_tensor_load_to_lds)
#    define HAS_TDM 1
#  endif
#endif
#ifndef HAS_TDM
#  define HAS_TDM 0
#endif

__device__ __forceinline__ unsigned short f2bf(float f) {
  unsigned int u = __builtin_bit_cast(unsigned int, f);
  u += 0x7FFFu + ((u >> 16) & 1u);           // round-to-nearest-even
  return (unsigned short)(u >> 16);
}
__device__ __forceinline__ float bf2f(unsigned short h) {
  unsigned int u = ((unsigned int)h) << 16;
  return __builtin_bit_cast(float, u);
}

// --------------------------------------------------------------------------
// Kernel 1a: bulk fp32 -> bf16 conversion of activations (vectorized, once).
// --------------------------------------------------------------------------
__global__ __launch_bounds__(256)
void xcvt_kernel(const float* __restrict__ X, unsigned short* __restrict__ Xb) {
  int t = (blockIdx.x * 256 + threadIdx.x) * 4;
  v4f x = *(const v4f*)(X + t);
  u16x4 o;
#pragma unroll
  for (int i = 0; i < 4; ++i) o[i] = f2bf(x[i]);
  *(u16x4*)(Xb + t) = o;
}

// --------------------------------------------------------------------------
// Kernel 1b: weights -> bf16, TRANSPOSED (WT[n][k]) so a WMMA B-fragment
// (fixed column n, contiguous K) is one contiguous 32-byte load.
// --------------------------------------------------------------------------
__global__ __launch_bounds__(256)
void wcvt_kernel(const float* __restrict__ Wq, const float* __restrict__ Wk,
                 const float* __restrict__ Wv, unsigned short* __restrict__ WqT,
                 unsigned short* __restrict__ WkT, unsigned short* __restrict__ WvT) {
  const float*    W  = (blockIdx.y == 0) ? Wq  : (blockIdx.y == 1) ? Wk  : Wv;
  unsigned short* WT = (blockIdx.y == 0) ? WqT : (blockIdx.y == 1) ? WkT : WvT;
  int t = blockIdx.x * 256 + threadIdx.x;     // 65536 elements / matrix
  int k = t >> 8;
  int n = t & 255;
  WT[n * 256 + k] = f2bf(W[k * 256 + n]);     // coalesced read, strided write
}

// --------------------------------------------------------------------------
// Kernel 2: Y_bf16[M,256] = relu(Xbf[M,256] @ W + bias) * scale
// 256 threads = 8 waves; block covers 32 rows x 256 cols; gridDim.y selects
// the (WT, bias, Y) set so the K and V projections share L2-hot X.
//
// The block's 32x256 bf16 X tile (16 KB) is DMA'd into LDS once by the
// Tensor Data Mover (D# per CDNA5 ISA ch.8), waited with s_wait_tensorcnt,
// then the k-loop reads A fragments with ds_load_b128 (zero redundant
// global A traffic) while B fragments stream from L2-resident weights.
//
// Fragment layouts used here (A and B share one contiguous k order):
//   A 16x32 bf16 : m = lane&15 ; elem j -> K = 16*(lane>>4) + j
//   B 32x16 bf16 : n = lane&15 ; elem j -> K = 16*(lane>>4) + j
//   C 16x16 f32  : n = lane&15 ; vgpr r -> M = r + 8*(lane>>4)
// Epilogue reuses the same LDS tile; global stores are contiguous b128.
// --------------------------------------------------------------------------
__global__ __launch_bounds__(256)
void proj_gemm_kernel(const unsigned short* __restrict__ Xbf,
                      const unsigned short* __restrict__ WT0,
                      const float* __restrict__ bias0,
                      unsigned short* __restrict__ Y0,
                      const unsigned short* __restrict__ WT1,
                      const float* __restrict__ bias1,
                      unsigned short* __restrict__ Y1,
                      float scale) {
  const unsigned short* WT   = blockIdx.y ? WT1   : WT0;
  const float*          bias = blockIdx.y ? bias1 : bias0;
  unsigned short*       Y    = blockIdx.y ? Y1    : Y0;

  __shared__ unsigned short stile[32 * H_SZ];   // 16 KB: X tile, then Y tile

  const int tid   = threadIdx.x;
  const int lane  = tid & 31;
  const int w     = tid >> 5;
  const int mloc  = (w >> 2) * 16;              // 0 or 16 (local M-tile)
  const int nbase = (w & 3) * 64;
  const int l15   = lane & 15;
  const int hi    = lane >> 4;                  // 0 or 1
  const int kbA   = hi * 16;
  const int kbB   = hi * 16;

  const unsigned short* xsrc = Xbf + (size_t)blockIdx.x * 32 * H_SZ;

  // ---- stage the 32x256 bf16 X tile into LDS ----
#if HAS_TDM
  if (w == 0) {                                  // one TDM issue per block
    unsigned lds_off = (unsigned)(unsigned long long)(void*)stile; // low 32b
    unsigned long long ga = (unsigned long long)xsrc;
    u32x4 g0;
    g0.x = 1u;                                   // count=1, user descriptor
    g0.y = lds_off;                              // D#.lds_addr
    g0.z = (unsigned)ga;                         // D#.global_addr[31:0]
    g0.w = (unsigned)(ga >> 32) | 0x80000000u;   // addr[56:32] | type=2
    i32x8 g1;
    g1[0] = 0x00010000;                          // data_size=1 (2 bytes)
    g1[1] = (int)(256u << 16);                   // tensor_dim0 = 256
    g1[2] = (int)(32u  << 16);                   // tensor_dim1 = 32
    g1[3] = (int)(256u << 16);                   // tile_dim0   = 256
    g1[4] = 32;                                  // tile_dim1   = 32
    g1[5] = 256;                                 // tensor_dim0_stride = 256
    g1[6] = 0;
    g1[7] = 0;
    i32x4 z4 = {};
    i32x8 z8 = {};
#  if __clang_major__ >= 23
    __builtin_amdgcn_tensor_load_to_lds(g0, g1, z4, z4, z8, 0);
#  else
    __builtin_amdgcn_tensor_load_to_lds(g0, g1, z4, z4, 0);
#  endif
    (void)z8;
    __builtin_amdgcn_s_wait_tensorcnt(0);
  }
#else
  // fallback: cooperative b128 copy (16 KB / block)
#pragma unroll
  for (int c = 0; c < 4; ++c) {
    int e = c * 2048 + tid * 8;
    *(u16x8*)(stile + e) = *(const u16x8*)(xsrc + e);
  }
#endif
  __syncthreads();

  const unsigned short* xrow = stile + (mloc + l15) * H_SZ;

  v8f acc0 = {}, acc1 = {}, acc2 = {}, acc3 = {};

#pragma unroll
  for (int kk = 0; kk < H_SZ; kk += 32) {
    // ---- A fragment from LDS: two ds_load_b128 ----
    u16x8 alo = *(const u16x8*)(xrow + kk + kbA);
    u16x8 ahi = *(const u16x8*)(xrow + kk + kbA + 8);
    u16x16 au = __builtin_shufflevector(alo, ahi,
                    0, 1, 2, 3, 4, 5, 6, 7, 8, 9, 10, 11, 12, 13, 14, 15);
    v16bf afrag = __builtin_bit_cast(v16bf, au);

    // ---- 4 B fragments (contiguous 32B each from transposed bf16 W) ----
#pragma unroll
    for (int t = 0; t < 4; ++t) {
      const int n = nbase + t * 16 + l15;
      u16x16 bu = *(const u16x16*)(WT + (size_t)n * H_SZ + kk + kbB);
      v16bf bfrag = __builtin_bit_cast(v16bf, bu);
      v8f c = (t == 0) ? acc0 : (t == 1) ? acc1 : (t == 2) ? acc2 : acc3;
      c = __builtin_amdgcn_wmma_f32_16x16x32_bf16(
              /*neg_a=*/false, afrag, /*neg_b=*/false, bfrag,
              /*c_mod=*/(short)0, c, /*reuse_a=*/false, /*reuse_b=*/false);
      if      (t == 0) acc0 = c;
      else if (t == 1) acc1 = c;
      else if (t == 2) acc2 = c;
      else             acc3 = c;
    }
  }

  __syncthreads();   // all LDS X reads done; reuse stile as the Y tile

  // ---- epilogue: +bias, relu, *scale -> LDS (scattered b16) ----
#pragma unroll
  for (int t = 0; t < 4; ++t) {
    const int n = nbase + t * 16 + l15;
    const float bb = bias[n];
    v8f c = (t == 0) ? acc0 : (t == 1) ? acc1 : (t == 2) ? acc2 : acc3;
#pragma unroll
    for (int r = 0; r < 8; ++r) {
      const int rowl = mloc + r + 8 * hi;
      float val = fmaxf(c[r] + bb, 0.0f) * scale;
      stile[rowl * H_SZ + n] = f2bf(val);
    }
  }
  __syncthreads();

  // ---- LDS -> global, fully coalesced b128 copies (16 KB / block) ----
  unsigned short* Yb = Y + (size_t)blockIdx.x * 32 * H_SZ;
#pragma unroll
  for (int c = 0; c < 4; ++c) {
    int e = c * 2048 + tid * 8;
    *(u16x8*)(Yb + e) = *(const u16x8*)(stile + e);
  }
}

// --------------------------------------------------------------------------
// Kernel 3: windowed attention. One block (256 thr = 8 waves) per query.
// Scores via wave32 shuffle reductions; softmax by wave 0; attn@V with
// coalesced bf16 reads. K/V rows are L2-resident (32 MB << 192 MB L2);
// global_prefetch hides the scattered-row latency.
// --------------------------------------------------------------------------
__global__ __launch_bounds__(256)
void attn_kernel(const unsigned short* __restrict__ Qb,
                 const unsigned short* __restrict__ Kb,
                 const unsigned short* __restrict__ Vb,
                 const int* __restrict__ starts, const int* __restrict__ ends,
                 float* __restrict__ out) {
  __shared__ float qs[H_SZ];
  __shared__ float sc[MAXW];
  __shared__ float invZ_s;

  const int b    = blockIdx.x;
  const int tid  = threadIdx.x;
  const int lane = tid & 31;
  const int w    = tid >> 5;

  const int start = starts[b];
  int len = ends[b] - start;
  len = (len < 1) ? 1 : (len > MAXW ? MAXW : len);

  // stage q (already relu'ed and pre-scaled by 1/16) as f32 in LDS
  qs[tid] = bf2f(Qb[(size_t)b * H_SZ + tid]);
  __syncthreads();

  // ---- scores: wave w handles neighbors j = w, w+8, ... ----
  for (int j = w; j < len; j += 8) {
    const unsigned short* krow = Kb + (size_t)(start + j) * H_SZ + lane * 8;
    if (j + 8 < len)
      __builtin_prefetch(krow + 8 * H_SZ, 0, 0);   // next K row for this wave
    u16x8 kv = *(const u16x8*)krow;                // 16B coalesced per lane
    float p = 0.0f;
#pragma unroll
    for (int i = 0; i < 8; ++i) p += qs[lane * 8 + i] * bf2f(kv[i]);
#pragma unroll
    for (int o = 16; o >= 1; o >>= 1) p += __shfl_xor(p, o, 32);
    if (lane == 0) sc[j] = p;
  }
  __syncthreads();

  // ---- softmax over the (<=64) window, done by wave 0 ----
  if (w == 0) {
    float s0 = (lane      < len) ? sc[lane]      : -1e30f;
    float s1 = (lane + 32 < len) ? sc[lane + 32] : -1e30f;
    float m = fmaxf(s0, s1);
#pragma unroll
    for (int o = 16; o >= 1; o >>= 1) m = fmaxf(m, __shfl_xor(m, o, 32));
    float e0 = (lane      < len) ? __expf(s0 - m) : 0.0f;
    float e1 = (lane + 32 < len) ? __expf(s1 - m) : 0.0f;
    sc[lane]      = e0;
    sc[lane + 32] = e1;
    float z = e0 + e1;
#pragma unroll
    for (int o = 16; o >= 1; o >>= 1) z += __shfl_xor(z, o, 32);
    if (lane == 0) invZ_s = 1.0f / z;
  }
  __syncthreads();

  // ---- out[h] = (sum_j e_j * V[start+j][h]) * invZ ; coalesced over h ----
  float acc = 0.0f;
  for (int j = 0; j < len; ++j) {
    const unsigned short* vrow = Vb + (size_t)(start + j) * H_SZ + tid;
    if (j + 4 < len)
      __builtin_prefetch(vrow + 4 * H_SZ, 0, 0);   // stream V rows ahead
    acc += sc[j] * bf2f(*vrow);
  }
  out[(size_t)b * H_SZ + tid] = acc * invZ_s;
}

// --------------------------------------------------------------------------
extern "C" void kernel_launch(void* const* d_in, const int* in_sizes, int n_in,
                              void* d_out, int out_size, void* d_ws, size_t ws_size,
                              hipStream_t stream) {
  const float* enc = (const float*)d_in[0];   // [B,H]
  const float* soc = (const float*)d_in[1];   // [N,H]
  const int*   st  = (const int*)d_in[2];     // [B]
  const int*   en  = (const int*)d_in[3];     // [B]
  const float* Wq  = (const float*)d_in[4];
  const float* bq  = (const float*)d_in[5];
  const float* Wk  = (const float*)d_in[6];
  const float* bk  = (const float*)d_in[7];
  const float* Wv  = (const float*)d_in[8];
  const float* bv  = (const float*)d_in[9];
  float* out = (float*)d_out;

  // workspace layout (bf16 = unsigned short), ~50 MB total
  unsigned short* ws   = (unsigned short*)d_ws;
  unsigned short* WqT  = ws;                        // 256*256
  unsigned short* WkT  = WqT + H_SZ * H_SZ;
  unsigned short* WvT  = WkT + H_SZ * H_SZ;
  unsigned short* Qb   = WvT + H_SZ * H_SZ;         // B*H
  unsigned short* Kb   = Qb  + (size_t)B_SZ * H_SZ; // N*H
  unsigned short* Vb   = Kb  + (size_t)N_SZ * H_SZ; // N*H
  unsigned short* EncB = Vb  + (size_t)N_SZ * H_SZ; // B*H  (bf16 enc_hidden)
  unsigned short* SocB = EncB + (size_t)B_SZ * H_SZ;// N*H  (bf16 social_ht)

  // 1) one-time bf16 conversions
  xcvt_kernel<<<(B_SZ * H_SZ) / 1024, 256, 0, stream>>>(enc, EncB);
  xcvt_kernel<<<(N_SZ * H_SZ) / 1024, 256, 0, stream>>>(soc, SocB);
  wcvt_kernel<<<dim3(H_SZ * H_SZ / 256, 3), 256, 0, stream>>>(
      Wq, Wk, Wv, WqT, WkT, WvT);

  // 2) projections (WMMA bf16 GEMMs)
  //    Q: small, scale = 1/sqrt(256) applied post-relu
  proj_gemm_kernel<<<dim3(B_SZ / 32, 1), 256, 0, stream>>>(
      EncB, WqT, bq, Qb, WqT, bq, Qb, 0.0625f);
  //    K and V fused in one launch (gridDim.y = 2) sharing L2-hot X
  proj_gemm_kernel<<<dim3(N_SZ / 32, 2), 256, 0, stream>>>(
      SocB, WkT, bk, Kb, WvT, bv, Vb, 1.0f);

  // 3) windowed attention
  attn_kernel<<<B_SZ, 256, 0, stream>>>(Qb, Kb, Vb, st, en, out);
}
